// DiffLogicPBF_69458211111035
// MI455X (gfx1250) — compile-verified, matrix operand order checked by reading, not run on hardware
//
#include <hip/hip_runtime.h>

#define BATCH   8192
#define KN      4096
#define NLAYERS 6
#define CLS     2

typedef __attribute__((ext_vector_type(16))) _Float16 v16h;
typedef __attribute__((ext_vector_type(8)))  float    v8f;

// LOGIC_COEFFS[f][c]: f_i(a,b) = c1 + ca*a + cb*b + cab*a*b
__constant__ float LOGIC_COEFFS[16][4] = {
    {0.f,  0.f,  0.f,  0.f}, {0.f,  0.f,  0.f,  1.f},
    {0.f,  1.f,  0.f, -1.f}, {0.f,  1.f,  0.f,  0.f},
    {0.f,  0.f,  1.f, -1.f}, {0.f,  0.f,  1.f,  0.f},
    {0.f,  1.f,  1.f, -2.f}, {0.f,  1.f,  1.f, -1.f},
    {1.f, -1.f, -1.f,  1.f}, {1.f, -1.f, -1.f,  2.f},
    {1.f,  0.f, -1.f,  0.f}, {1.f,  0.f, -1.f,  1.f},
    {1.f, -1.f,  0.f,  0.f}, {1.f, -1.f,  0.f,  1.f},
    {1.f,  0.f,  0.f, -1.f}, {1.f,  0.f,  0.f,  0.f}};

__device__ __forceinline__ float bflo(unsigned int u) {
    union { unsigned int i; float f; } v; v.i = u << 16; return v.f;
}
__device__ __forceinline__ float bfhi(unsigned int u) {
    union { unsigned int i; float f; } v; v.i = u & 0xFFFF0000u; return v.f;
}
__device__ __forceinline__ unsigned int f2bf(float f) {
    union { float f; unsigned int i; } v; v.f = f;
    return (v.i + 0x7FFFu + ((v.i >> 16) & 1u)) >> 16;  // RNE
}

// ---------------------------------------------------------------------------
// Coefficient precompute: c[layer][k][0..3] = softmax(w[k,:]) @ LOGIC_COEFFS
// One wave32 per 16-neuron tile; the 16x16 @ 16x4 (zero-padded to 16) product
// runs on v_wmma_f32_16x16x32_f16 (K padded 16 -> 32 with zeros).
// A layout (16-bit A, 16x32): lane<16 holds row M=lane, elems 0..7 = K0..7;
// lane>=16 holds row M=lane-16, elems 0..7 = K8..15; elems 8..15 (K>=16) = 0.
// B layout mirrored: lane<16 = col N=lane, elems 0..7 = K0..7; hi half K8..15.
// D layout: lane L holds col N=L%16; vgpr r holds row M = r + 8*(L>=16).
// ---------------------------------------------------------------------------
__global__ void dl_coef_kernel(const float* __restrict__ w0,
                               const float* __restrict__ wrest,
                               float* __restrict__ coefs) {
    const int tile  = blockIdx.x;            // 0 .. NLAYERS*(KN/16)-1
    const int layer = tile / (KN / 16);
    const int t     = tile % (KN / 16);
    const float* w  = (layer == 0) ? w0 : (wrest + (size_t)(layer - 1) * KN * 16);

    const int  lane = threadIdx.x;           // 0..31, full wave, no divergence
    const int  row  = lane & 15;
    const bool lo   = lane < 16;
    const float* wr = w + (size_t)(t * 16 + row) * 16;

    // softmax over 16 logic functions (each half-wave lane redoes its row: cheap)
    float vals[16];
    float m = -3.4e38f;
#pragma unroll
    for (int i = 0; i < 16; ++i) { vals[i] = wr[i]; m = fmaxf(m, vals[i]); }
    float s = 0.f;
#pragma unroll
    for (int i = 0; i < 16; ++i) { vals[i] = __expf(vals[i] - m); s += vals[i]; }
    const float inv = 1.0f / s;

    v16h a;
#pragma unroll
    for (int e = 0; e < 8; ++e) {
        float vlo = vals[e] * inv;
        float vhi = vals[8 + e] * inv;
        a[e] = (_Float16)(lo ? vlo : vhi);
    }
#pragma unroll
    for (int e = 8; e < 16; ++e) a[e] = (_Float16)0.f;   // K = 16..31 padding

    const int n = lane & 15;                 // output column (coef index)
    v16h b;
#pragma unroll
    for (int e = 0; e < 8; ++e) {
        float blo2 = LOGIC_COEFFS[e][n & 3];
        float bhi2 = LOGIC_COEFFS[8 + e][n & 3];
        float bv   = lo ? blo2 : bhi2;
        b[e] = (n < 4) ? (_Float16)bv : (_Float16)0.f;
    }
#pragma unroll
    for (int e = 8; e < 16; ++e) b[e] = (_Float16)0.f;

    v8f c = {};
    c = __builtin_amdgcn_wmma_f32_16x16x32_f16(
            /*neg_a=*/false, a, /*neg_b=*/false, b,
            /*c_mod=*/(short)0, c, /*reuse_a=*/false, /*reuse_b=*/false);

    if (n < 4) {
        const int mbase = lo ? 0 : 8;
#pragma unroll
        for (int r = 0; r < 8; ++r) {
            coefs[(size_t)(layer * KN + t * 16 + mbase + r) * 4 + n] = c[r];
        }
    }
}

// ---------------------------------------------------------------------------
// Layer 0: binarize x[B,2], apply 2->K logic layer. h layout is [K, B] bf16.
// ---------------------------------------------------------------------------
__global__ void dl_layer0_kernel(const float* __restrict__ x,
                                 const int* __restrict__ idx0,
                                 const float* __restrict__ coefs,
                                 unsigned short* __restrict__ hout) {
    const int k  = blockIdx.y;
    const int ia = idx0[k];
    const int ib = idx0[KN + k];
    const float4 cf = reinterpret_cast<const float4*>(coefs)[k];
    const int b = blockIdx.x * blockDim.x + threadIdx.x;
    const float av = x[b * 2 + ia] > 0.0f ? 1.0f : 0.0f;
    const float bv = x[b * 2 + ib] > 0.0f ? 1.0f : 0.0f;
    const float r  = fmaf(av, fmaf(bv, cf.w, cf.y), fmaf(bv, cf.z, cf.x));
    hout[(size_t)k * BATCH + b] = (unsigned short)f2bf(r);
}

// ---------------------------------------------------------------------------
// Logic layer K->K. [K,B] layout => gather of column idx becomes a contiguous
// row read. Per-block-uniform indices/coefs (scalar path); 128-bit bf16 loads.
// Each thread handles 8 batch elements.
// ---------------------------------------------------------------------------
__global__ void dl_layer_kernel(const unsigned short* __restrict__ hprev,
                                unsigned short* __restrict__ hout,
                                const int* __restrict__ idx,
                                const float* __restrict__ coefs) {
    const int k  = blockIdx.y;
    const int ia = idx[k];
    const int ib = idx[KN + k];
    const float4 cf = reinterpret_cast<const float4*>(coefs)[k];
    const int b0 = (blockIdx.x * blockDim.x + threadIdx.x) * 8;

    const uint4 ua = *reinterpret_cast<const uint4*>(hprev + (size_t)ia * BATCH + b0);
    const uint4 ub = *reinterpret_cast<const uint4*>(hprev + (size_t)ib * BATCH + b0);

    auto ev = [&](unsigned int wa, unsigned int wb) -> unsigned int {
        const float a0 = bflo(wa), a1 = bfhi(wa);
        const float c0 = bflo(wb), c1 = bfhi(wb);
        const float r0 = fmaf(a0, fmaf(c0, cf.w, cf.y), fmaf(c0, cf.z, cf.x));
        const float r1 = fmaf(a1, fmaf(c1, cf.w, cf.y), fmaf(c1, cf.z, cf.x));
        return f2bf(r0) | (f2bf(r1) << 16);
    };

    uint4 uo;
    uo.x = ev(ua.x, ub.x);
    uo.y = ev(ua.y, ub.y);
    uo.z = ev(ua.z, ub.z);
    uo.w = ev(ua.w, ub.w);
    *reinterpret_cast<uint4*>(hout + (size_t)k * BATCH + b0) = uo;
}

// ---------------------------------------------------------------------------
// GroupSum, two deterministic stages (no float atomics -> bit-stable replays).
// Stage 1: 16-way K-split partial sums. Stage 2: reduce 16 partials.
// ---------------------------------------------------------------------------
__global__ void dl_gsum_partial(const unsigned short* __restrict__ h,
                                float* __restrict__ partial) {
    const int b   = blockIdx.x * blockDim.x + threadIdx.x;
    const int cls = blockIdx.y;
    const int s   = blockIdx.z;
    const int kspan = (KN / CLS) / 16;           // 128
    const int k0    = cls * (KN / CLS) + s * kspan;
    float acc = 0.f;
#pragma unroll 4
    for (int j = 0; j < kspan; ++j) {
        acc += bflo((unsigned int)h[(size_t)(k0 + j) * BATCH + b]);
    }
    partial[(size_t)(cls * 16 + s) * BATCH + b] = acc;
}

__global__ void dl_gsum_final(const float* __restrict__ partial,
                              float* __restrict__ out) {
    const int b   = blockIdx.x * blockDim.x + threadIdx.x;
    const int cls = blockIdx.y;
    float acc = 0.f;
#pragma unroll
    for (int s = 0; s < 16; ++s)
        acc += partial[(size_t)(cls * 16 + s) * BATCH + b];
    out[b * CLS + cls] = acc;
}

// ---------------------------------------------------------------------------
extern "C" void kernel_launch(void* const* d_in, const int* in_sizes, int n_in,
                              void* d_out, int out_size, void* d_ws, size_t ws_size,
                              hipStream_t stream) {
    (void)in_sizes; (void)n_in; (void)out_size; (void)ws_size;

    const float* x    = (const float*)d_in[0];   // [8192, 2]
    const float* w0   = (const float*)d_in[1];   // [4096, 16]
    const float* wsW  = (const float*)d_in[2];   // [5, 4096, 16]
    const int*   idx0 = (const int*)d_in[3];     // [2, 4096]
    const int*   idxs = (const int*)d_in[4];     // [5, 2, 4096]
    float* out = (float*)d_out;                  // [8192, 2] f32

    // Scratch layout: coefs | hA (64MB) | hB (64MB) | partials (1MB)
    char* wsb = (char*)d_ws;
    float* coefs = (float*)wsb;                                  // L*K*4 f32
    const size_t coefBytes = (size_t)NLAYERS * KN * 4 * sizeof(float);
    unsigned short* hA = (unsigned short*)(wsb + coefBytes);
    unsigned short* hB = hA + (size_t)KN * BATCH;
    float* partial = (float*)(hB + (size_t)KN * BATCH);

    // 1) per-neuron coefficients via softmax + WMMA (1 wave per 16 neurons)
    dl_coef_kernel<<<NLAYERS * (KN / 16), 32, 0, stream>>>(w0, wsW, coefs);

    // 2) layer 0 (binarize + 2->K)
    dl_layer0_kernel<<<dim3(BATCH / 256, KN), 256, 0, stream>>>(x, idx0, coefs, hA);

    // 3) layers 1..5, ping-pong between the two L2-resident bf16 buffers
    unsigned short* src = hA;
    unsigned short* dst = hB;
    for (int l = 1; l < NLAYERS; ++l) {
        dl_layer_kernel<<<dim3(BATCH / (256 * 8), KN), 256, 0, stream>>>(
            src, dst, idxs + (size_t)(l - 1) * 2 * KN, coefs + (size_t)l * KN * 4);
        unsigned short* tmp = src; src = dst; dst = tmp;
    }
    // final activations now in `src`

    // 4) GroupSum (deterministic 2-stage)
    dl_gsum_partial<<<dim3(BATCH / 256, CLS, 16), 256, 0, stream>>>(src, partial);
    dl_gsum_final<<<dim3(BATCH / 256, CLS), 256, 0, stream>>>(partial, out);
}
